// ContrastiveLoss_79869211837075
// MI455X (gfx1250) — matrix operand order repaired and checked by reading, not compiled, not run on hardware
//
#include <hip/hip_runtime.h>
#include <hip/hip_bf16.h>
#include <stdint.h>

// ---------------------------------------------------------------------------
// ContrastiveLoss on MI455X (gfx1250, wave32, WMMA)
//   d^2 = ||a||^2 + ||b||^2 - 2 a.b  via bf16 split-precision WMMA gram matmul
//   - register double-buffered K pipeline (global loads overlap 24 WMMAs)
//   - tile-wide global_prefetch of labels/weights overlaps the 512 MB HBM
//     stream (the roofline floor: ~22 us @ 23.3 TB/s) with the GEMM phase
//   - fused epilogue + deterministic two-stage reduction
// ---------------------------------------------------------------------------

typedef __attribute__((ext_vector_type(16))) __bf16 v16bf;
typedef __attribute__((ext_vector_type(8)))  float  v8f;

#define NN 8192
#define DD 512
#define TM 128          // workgroup tile rows
#define TN 128          // workgroup tile cols
#define KC 32           // K chunk = one bf16 WMMA k-step
#define LDT 40          // padded LDS row stride in ushorts (80 B, 16B aligned)
#define NWG_X (NN / TM) // 64
#define NWG (NWG_X * NWG_X)

__device__ __forceinline__ unsigned short f2bf(float f) {
    unsigned u = __float_as_uint(f);
    unsigned r = (u + 0x7FFFu + ((u >> 16) & 1u)) >> 16; // RNE truncate
    return (unsigned short)r;
}
__device__ __forceinline__ float bf2f(unsigned short b) {
    return __uint_as_float(((unsigned)b) << 16);
}

union FragU { uint4 q[2]; v16bf v; };

// Build a 16 x bf16 fragment (8 VGPRs) from two 16-byte LDS chunks.
__device__ __forceinline__ v16bf ld_frag(const unsigned short* p0,
                                         const unsigned short* p1) {
    FragU f;
    f.q[0] = *(const uint4*)p0;  // ds_load_b128
    f.q[1] = *(const uint4*)p1;  // ds_load_b128
    return f.v;
}

// ---------------------------------------------------------------------------
// Kernel 1: fp32 squared norms per row
// ---------------------------------------------------------------------------
__global__ void ContrastiveLoss_norms_kernel(const float* __restrict__ emb,
                                             float* __restrict__ norms) {
    __shared__ float red[256];
    const int row = blockIdx.x;
    const float* r = emb + (size_t)row * DD;
    float s = 0.f;
    for (int k = threadIdx.x; k < DD; k += 256) {
        float v = r[k];
        s += v * v;
    }
    red[threadIdx.x] = s;
    __syncthreads();
    for (int off = 128; off > 0; off >>= 1) {
        if ((int)threadIdx.x < off) red[threadIdx.x] += red[threadIdx.x + off];
        __syncthreads();
    }
    if (threadIdx.x == 0) norms[row] = red[0];
}

// ---------------------------------------------------------------------------
// Tile pipeline helpers
// ---------------------------------------------------------------------------
__device__ __forceinline__ void tile_load_regs(const float* __restrict__ emb,
                                               int rowBase, int colBase, int k0,
                                               int tid, float4 ra[4], float4 rb[4]) {
#pragma unroll
    for (int it = 0; it < 4; ++it) {
        const int ch  = tid + it * 256;      // 0..1023 : 128 rows x 8 chunks
        const int row = ch >> 3;
        const int kp  = ch & 7;
        ra[it] = *(const float4*)(emb + (size_t)(rowBase + row) * DD + k0 + kp * 4);
        rb[it] = *(const float4*)(emb + (size_t)(colBase + row) * DD + k0 + kp * 4);
    }
}

__device__ __forceinline__ void tile_cvt_store(unsigned short* sAhi, unsigned short* sAlo,
                                               unsigned short* sBhi, unsigned short* sBlo,
                                               int tid, const float4 ra[4], const float4 rb[4]) {
#pragma unroll
    for (int it = 0; it < 4; ++it) {
        const int ch  = tid + it * 256;
        const int row = ch >> 3;
        const int kp  = ch & 7;
        const float4 va = ra[it];
        const float4 vb = rb[it];

        unsigned short ah0 = f2bf(va.x), ah1 = f2bf(va.y), ah2 = f2bf(va.z), ah3 = f2bf(va.w);
        unsigned short al0 = f2bf(va.x - bf2f(ah0)), al1 = f2bf(va.y - bf2f(ah1));
        unsigned short al2 = f2bf(va.z - bf2f(ah2)), al3 = f2bf(va.w - bf2f(ah3));
        unsigned short bh0 = f2bf(vb.x), bh1 = f2bf(vb.y), bh2 = f2bf(vb.z), bh3 = f2bf(vb.w);
        unsigned short bl0 = f2bf(vb.x - bf2f(bh0)), bl1 = f2bf(vb.y - bf2f(bh1));
        unsigned short bl2 = f2bf(vb.z - bf2f(bh2)), bl3 = f2bf(vb.w - bf2f(bh3));

        uint2 p;
        p.x = (unsigned)ah0 | ((unsigned)ah1 << 16);
        p.y = (unsigned)ah2 | ((unsigned)ah3 << 16);
        *(uint2*)&sAhi[row * LDT + kp * 4] = p;
        p.x = (unsigned)al0 | ((unsigned)al1 << 16);
        p.y = (unsigned)al2 | ((unsigned)al3 << 16);
        *(uint2*)&sAlo[row * LDT + kp * 4] = p;
        p.x = (unsigned)bh0 | ((unsigned)bh1 << 16);
        p.y = (unsigned)bh2 | ((unsigned)bh3 << 16);
        *(uint2*)&sBhi[row * LDT + kp * 4] = p;
        p.x = (unsigned)bl0 | ((unsigned)bl1 << 16);
        p.y = (unsigned)bl2 | ((unsigned)bl3 << 16);
        *(uint2*)&sBlo[row * LDT + kp * 4] = p;
    }
}

// ---------------------------------------------------------------------------
// Kernel 2: tiled split-bf16 WMMA gram matmul + fused loss epilogue
//   block = 256 threads = 8 waves; tile 128x128; wave sub-tile 32x64
// ---------------------------------------------------------------------------
__global__ __launch_bounds__(256)
void ContrastiveLoss_gram_kernel(const float* __restrict__ emb,
                                 const int*   __restrict__ labels,
                                 const float* __restrict__ weights,
                                 const float* __restrict__ norms,
                                 float2*      __restrict__ partials) {
    __shared__ __align__(16) unsigned short sAhi[TM * LDT];
    __shared__ __align__(16) unsigned short sAlo[TM * LDT];
    __shared__ __align__(16) unsigned short sBhi[TN * LDT];
    __shared__ __align__(16) unsigned short sBlo[TN * LDT];
    __shared__ float sNR[TM], sNC[TN];
    __shared__ float sRs[256], sRc[256];

    const int tid  = threadIdx.x;
    const int lane = tid & 31;
    const int wid  = tid >> 5;
    const int half = lane >> 4;      // 0: lanes 0-15, 1: lanes 16-31
    const int l15  = lane & 15;
    const int rowBase = blockIdx.y * TM;
    const int colBase = blockIdx.x * TN;

    // ---- warm L2 with this tile's label/weight stream (global_prefetch) ----
    // 128 rows x 512 B per array; 4 threads/row -> 128 B (one line) each.
#pragma unroll
    for (int rr = 0; rr < 2; ++rr) {
        const int r   = (tid >> 2) + rr * 64;
        const int seg = (tid & 3) * 32;
        const size_t off = (size_t)(rowBase + r) * NN + (size_t)(colBase + seg);
        __builtin_prefetch(labels + off, 0, 1);
        __builtin_prefetch(weights + off, 0, 1);
    }

    // stage per-tile norms
    if (tid < TM) sNR[tid] = norms[rowBase + tid];
    else          sNC[tid - TM] = norms[colBase + (tid - TM)];

    const int waveM = wid >> 1;  // 0..3 -> 32-row band
    const int waveN = wid & 1;   // 0..1 -> 64-col band

    v8f acc[2][4];
#pragma unroll
    for (int i = 0; i < 2; ++i)
#pragma unroll
        for (int j = 0; j < 4; ++j) acc[i][j] = v8f{};

    // ---- software pipeline: prologue loads chunk 0 ------------------------
    float4 ra[4], rb[4];
    tile_load_regs(emb, rowBase, colBase, 0, tid, ra, rb);
    tile_cvt_store(sAhi, sAlo, sBhi, sBlo, tid, ra, rb);
    __syncthreads();

    for (int k0 = KC; k0 <= DD; k0 += KC) {
        // issue next chunk's global loads before the WMMA block
        if (k0 < DD) tile_load_regs(emb, rowBase, colBase, k0, tid, ra, rb);

        // -------- A fragments: 16x32 bf16 layout (ISA 7.12.2) --------------
        v16bf ahi[2], alo[2];
#pragma unroll
        for (int rt = 0; rt < 2; ++rt) {
            const int r  = waveM * 32 + rt * 16 + l15;
            const int ko = half * 8;             // lanes16-31: K=8..15 / 24..31
            const unsigned short* p = &sAhi[r * LDT + ko];
            ahi[rt] = ld_frag(p, p + 16);
            const unsigned short* q = &sAlo[r * LDT + ko];
            alo[rt] = ld_frag(q, q + 16);
        }
        // -------- B fragments + 24 WMMAs -----------------------------------
#pragma unroll
        for (int ct = 0; ct < 4; ++ct) {
            const int c  = waveN * 64 + ct * 16 + l15;
            const int ko = half * 8;             // lanes16-31: K=8..15 / 24..31
            const unsigned short* pb = &sBhi[c * LDT + ko];
            const v16bf bhi = ld_frag(pb, pb + 16);
            const unsigned short* qb = &sBlo[c * LDT + ko];
            const v16bf blo = ld_frag(qb, qb + 16);
#pragma unroll
            for (int rt = 0; rt < 2; ++rt) {
                acc[rt][ct] = __builtin_amdgcn_wmma_f32_16x16x32_bf16(
                    false, ahi[rt], false, bhi, (short)0, acc[rt][ct], false, false);
                acc[rt][ct] = __builtin_amdgcn_wmma_f32_16x16x32_bf16(
                    false, ahi[rt], false, blo, (short)0, acc[rt][ct], false, false);
                acc[rt][ct] = __builtin_amdgcn_wmma_f32_16x16x32_bf16(
                    false, alo[rt], false, bhi, (short)0, acc[rt][ct], false, false);
            }
        }

        if (k0 < DD) {
            __syncthreads();   // all waves done reading current LDS tile
            tile_cvt_store(sAhi, sAlo, sBhi, sBlo, tid, ra, rb);
            __syncthreads();   // next tile visible to all waves
        }
    }

    // -------- fused epilogue: distances -> masked hinge loss ---------------
    float lsum = 0.f, lcnt = 0.f;
#pragma unroll
    for (int rt = 0; rt < 2; ++rt) {
#pragma unroll
        for (int ct = 0; ct < 4; ++ct) {
#pragma unroll
            for (int i = 0; i < 8; ++i) {
                // C/D layout: VGPR i -> M = i + 8*half, N = lane&15
                const int rl = waveM * 32 + rt * 16 + half * 8 + i;
                const int cl = waveN * 64 + ct * 16 + l15;
                float d2 = sNR[rl] + sNC[cl] - 2.0f * acc[rt][ct][i];
                d2 = fmaxf(d2, 1e-12f);
                const size_t idx = (size_t)(rowBase + rl) * NN + (size_t)(colBase + cl);
                const int   lbl = labels[idx];
                const float w   = weights[idx];
                const float d   = sqrtf(d2);
                const float h   = fmaxf(1.0f - d, 0.0f);
                const float isPos = (lbl == 1) ? 1.0f : 0.0f;
                const float isNeg = (lbl == 0) ? 1.0f : 0.0f;
                lsum += (isPos * d2 + isNeg * h * h) * w;
                lcnt += isPos + isNeg;
            }
        }
    }

    // -------- deterministic block reduction --------------------------------
    sRs[tid] = lsum;
    sRc[tid] = lcnt;
    __syncthreads();
    for (int off = 128; off > 0; off >>= 1) {
        if (tid < off) {
            sRs[tid] += sRs[tid + off];
            sRc[tid] += sRc[tid + off];
        }
        __syncthreads();
    }
    if (tid == 0)
        partials[blockIdx.y * gridDim.x + blockIdx.x] = make_float2(sRs[0], sRc[0]);
}

// ---------------------------------------------------------------------------
// Kernel 3: fixed-order final reduction
// ---------------------------------------------------------------------------
__global__ void ContrastiveLoss_final_kernel(const float2* __restrict__ partials,
                                             float* __restrict__ out) {
    __shared__ float ss[256], sc[256];
    float s = 0.f, c = 0.f;
    for (int i = threadIdx.x; i < NWG; i += 256) {
        const float2 p = partials[i];
        s += p.x;
        c += p.y;
    }
    ss[threadIdx.x] = s;
    sc[threadIdx.x] = c;
    __syncthreads();
    for (int off = 128; off > 0; off >>= 1) {
        if ((int)threadIdx.x < off) {
            ss[threadIdx.x] += ss[threadIdx.x + off];
            sc[threadIdx.x] += sc[threadIdx.x + off];
        }
        __syncthreads();
    }
    if (threadIdx.x == 0) out[0] = ss[0] / fmaxf(sc[0], 1.0f);
}

// ---------------------------------------------------------------------------
extern "C" void kernel_launch(void* const* d_in, const int* in_sizes, int n_in,
                              void* d_out, int out_size, void* d_ws, size_t ws_size,
                              hipStream_t stream) {
    const float* emb     = (const float*)d_in[0];
    const int*   labels  = (const int*)d_in[1];
    const float* weights = (const float*)d_in[2];

    float*  norms    = (float*)d_ws;                                        // 32 KB
    float2* partials = (float2*)((char*)d_ws + (size_t)NN * sizeof(float)); // 32 KB

    ContrastiveLoss_norms_kernel<<<NN, 256, 0, stream>>>(emb, norms);

    dim3 grid(NWG_X, NWG_X);
    ContrastiveLoss_gram_kernel<<<grid, 256, 0, stream>>>(emb, labels, weights,
                                                          norms, partials);

    ContrastiveLoss_final_kernel<<<1, 256, 0, stream>>>(partials, (float*)d_out);
}
